// GraphLearn_88725434401459
// MI455X (gfx1250) — matrix operand order repaired and checked by reading, not compiled, not run on hardware
//
#include <hip/hip_runtime.h>

typedef __attribute__((ext_vector_type(2)))  float  v2f;
typedef __attribute__((ext_vector_type(8)))  float  v8f;
typedef __attribute__((ext_vector_type(16))) __bf16 v16bf;
typedef __attribute__((ext_vector_type(8)))  __bf16 v8bf;

static constexpr int NN   = 6144;  // rows
static constexpr int DD   = 256;   // x proj dim
static constexpr int DO   = 600;   // origin proj dim
static constexpr int DOP  = 640;   // padded origin dim
static constexpr int KZ   = 896;   // 256 + 640 combined K
static constexpr int KTOP = 30;
static constexpr int TT   = NN / 128;            // 48 tile rows/cols
static constexpr int NTRI = TT * (TT + 1) / 2;   // 1176 upper-tri tiles

union V16U { v16bf v; v8bf h8[2]; };

// LDS byte offset of a __shared__ pointer: flat LDS aperture keeps the offset
// in addr[31:0] (ISA 10.2 aperture rules), so truncation recovers it.
__device__ __forceinline__ uint32_t lds_off(const void* p) {
  return (uint32_t)(uint64_t)(uintptr_t)p;
}

// Async 16B global -> LDS copy (per-lane), tracked by ASYNCcnt.
__device__ __forceinline__ void async_b128(uint32_t lds, uint64_t gaddr) {
  asm volatile("global_load_async_to_lds_b128 %0, %1, off"
               :: "v"(lds), "v"(gaddr) : "memory");
}
__device__ __forceinline__ void wait_async0() {
  asm volatile("s_wait_asynccnt 0" ::: "memory");
}

// ---------------------------------------------------------------------------
// Kernel 1: Z[:,0:256] = sqrt(0.9) * normalize(x @ p_w^T + p_b), split bf16 hi/lo
// ---------------------------------------------------------------------------
__global__ __launch_bounds__(512) void proj_x_kernel(
    const float* __restrict__ x, const float* __restrict__ pw,
    const float* __restrict__ pb,
    __bf16* __restrict__ Zh, __bf16* __restrict__ Zl)
{
  __shared__ float sh[16 * DD];
  __shared__ float ps[256];
  __shared__ float rscale[16];

  const int tid  = threadIdx.x;
  const int lane = tid & 31;
  const int wave = tid >> 5;
  const int l15  = lane & 15;
  const int h    = lane >> 4;
  const int rowbase = blockIdx.x * 16;

  const int arow = rowbase + l15;      // A row (M = l15)
  const int bcol = wave * 16 + l15;    // output col / p_w row (N = l15)

  v8f acc = {};
  const float* aptr = x  + (size_t)arow * DD + h * 2;
  const float* bptr = pw + (size_t)bcol * DD + h * 2;
  for (int k0 = 0; k0 < DD; k0 += 4) {
    v2f a = *(const v2f*)(aptr + k0);   // A 16x4 f32: lane row M, K = 2h,2h+1
    v2f b = *(const v2f*)(bptr + k0);   // B 4x16 f32: lane col N, K = 2h,2h+1
    acc = __builtin_amdgcn_wmma_f32_16x16x4_f32(false, a, false, b,
                                                (short)0, acc, false, false);
  }
  const float bias = pb[bcol];
#pragma unroll
  for (int v = 0; v < 8; ++v) {
    int m = v + h * 8;                 // C layout: VGPR v -> M=v (+8 hi lanes)
    sh[m * DD + bcol] = acc[v] + bias;
  }
  __syncthreads();

  if (tid < 256) {
    int row = tid >> 4, seg = tid & 15;
    float s = 0.f;
#pragma unroll
    for (int c = 0; c < 16; ++c) { float v = sh[row * DD + seg * 16 + c]; s += v * v; }
    ps[tid] = s;
  }
  __syncthreads();
  if (tid < 16) {
    float s = 0.f;
#pragma unroll
    for (int c = 0; c < 16; ++c) s += ps[tid * 16 + c];
    rscale[tid] = 0.948683298050514f / fmaxf(sqrtf(s), 1e-12f);  // sqrt(0.9)/||v||
  }
  __syncthreads();

  for (int idx = tid; idx < 16 * DD; idx += 512) {
    int row = idx >> 8, col = idx & 255;
    float v = sh[row * DD + col] * rscale[row];
    __bf16 hi = (__bf16)v;
    __bf16 lo = (__bf16)(v - (float)hi);
    size_t off = (size_t)(rowbase + row) * KZ + col;
    Zh[off] = hi; Zl[off] = lo;
  }
}

// ---------------------------------------------------------------------------
// Kernel 2: Z[:,256:896] = sqrt(0.1*c_j) * normalize(x_origin @ po_w^T + po_b),
// padded 600->640 with zeros, split bf16 hi/lo.
// ---------------------------------------------------------------------------
__global__ __launch_bounds__(512) void proj_o_kernel(
    const float* __restrict__ xo, const float* __restrict__ pow_,
    const float* __restrict__ pob,
    __bf16* __restrict__ Zh, __bf16* __restrict__ Zl)
{
  __shared__ float sh[16 * 608];
  __shared__ float ps[256];
  __shared__ float rscale[16];

  const int tid  = threadIdx.x;
  const int lane = tid & 31;
  const int wave = tid >> 5;
  const int l15  = lane & 15;
  const int h    = lane >> 4;
  const int rowbase = blockIdx.x * 16;
  const int arow = rowbase + l15;

  const int  ct0 = wave, ct1 = wave + 16, ct2 = wave + 32;
  const bool has2 = (ct2 < 38);
  const int  ct2c = has2 ? ct2 : ct0;
  const int  c0 = ct0 * 16 + l15, c1 = ct1 * 16 + l15, c2 = ct2c * 16 + l15;
  const int  w0 = (c0 < DO) ? c0 : 0;
  const int  w1 = (c1 < DO) ? c1 : 0;
  const int  w2 = (c2 < DO) ? c2 : 0;

  v8f a0 = {}, a1 = {}, a2 = {};
  const float* aptr = xo   + (size_t)arow * DO + h * 2;
  const float* b0p  = pow_ + (size_t)w0 * DO + h * 2;
  const float* b1p  = pow_ + (size_t)w1 * DO + h * 2;
  const float* b2p  = pow_ + (size_t)w2 * DO + h * 2;
  for (int k0 = 0; k0 < DO; k0 += 4) {      // 600 % 4 == 0
    v2f a  = *(const v2f*)(aptr + k0);
    v2f b0 = *(const v2f*)(b0p + k0);
    v2f b1 = *(const v2f*)(b1p + k0);
    v2f b2 = *(const v2f*)(b2p + k0);
    a0 = __builtin_amdgcn_wmma_f32_16x16x4_f32(false, a, false, b0, (short)0, a0, false, false);
    a1 = __builtin_amdgcn_wmma_f32_16x16x4_f32(false, a, false, b1, (short)0, a1, false, false);
    a2 = __builtin_amdgcn_wmma_f32_16x16x4_f32(false, a, false, b2, (short)0, a2, false, false);
  }
#pragma unroll
  for (int v = 0; v < 8; ++v) {
    int m = v + h * 8;
    sh[m * 608 + c0] = (c0 < DO) ? (a0[v] + pob[c0]) : 0.f;
    sh[m * 608 + c1] = (c1 < DO) ? (a1[v] + pob[c1]) : 0.f;
    if (has2) sh[m * 608 + c2] = (c2 < DO) ? (a2[v] + pob[c2]) : 0.f;
  }
  __syncthreads();

  if (tid < 256) {
    int row = tid >> 4, seg = tid & 15;
    int cbeg = seg * 38, cend = cbeg + 38; if (cend > DO) cend = DO;
    float s = 0.f;
    for (int c = cbeg; c < cend; ++c) { float v = sh[row * 608 + c]; s += v * v; }
    ps[tid] = s;
  }
  __syncthreads();
  if (tid < 16) {
    float s = 0.f;
#pragma unroll
    for (int c = 0; c < 16; ++c) s += ps[tid * 16 + c];
    rscale[tid] = 1.0f / fmaxf(sqrtf(s), 1e-12f);
  }
  __syncthreads();

  const float coef[6] = {0.17f, 0.5f, 0.14f, 0.16f, 0.9f, 0.15f};
  for (int idx = tid; idx < 16 * DOP; idx += 512) {
    int row = idx / DOP, j = idx - row * DOP;
    float v = 0.f;
    if (j < DO) v = sh[row * 608 + j] * rscale[row] * sqrtf(0.1f * coef[j / 100]);
    __bf16 hi = (__bf16)v;
    __bf16 lo = (__bf16)(v - (float)hi);
    size_t off = (size_t)(rowbase + row) * KZ + 256 + j;
    Zh[off] = hi; Zl[off] = lo;
  }
}

// ---------------------------------------------------------------------------
// Kernel 3: scores = Z @ Z^T, bf16 hi/lo split (3 WMMA products, f32 accum).
// Upper-triangle 128x128 tiles only (symmetry); off-diagonal tiles also store
// the transpose (contiguous b128 stores thanks to the C fragment layout).
// Double-buffered LDS fed by GLOBAL_LOAD_ASYNC_TO_LDS_B128 (ASYNCcnt).
// LDS rows padded to 40 bf16 -> conflict-free b128 fragment reads.
// ---------------------------------------------------------------------------
__global__ __launch_bounds__(512) void gram_kernel(
    const __bf16* __restrict__ Zh, const __bf16* __restrict__ Zl,
    float* __restrict__ S)
{
  __shared__ __align__(16) __bf16 Ah[2][128 * 40];
  __shared__ __align__(16) __bf16 Al[2][128 * 40];
  __shared__ __align__(16) __bf16 Bh[2][128 * 40];
  __shared__ __align__(16) __bf16 Bl[2][128 * 40];

  // ---- decode upper-triangle tile (by <= bx) from linear block id ----
  const int t = blockIdx.x;
  int by = (int)((2 * TT + 1 - sqrtf((float)((2 * TT + 1) * (2 * TT + 1) - 8 * t))) * 0.5f);
  if (by < 0) by = 0;
  if (by > TT - 1) by = TT - 1;
  while (by > 0 && (by * (2 * TT + 1 - by)) / 2 > t) --by;
  while (((by + 1) * (2 * TT + 1 - (by + 1))) / 2 <= t) ++by;
  const int bx = by + (t - (by * (2 * TT + 1 - by)) / 2);

  const int tid  = threadIdx.x;
  const int lane = tid & 31;
  const int wave = tid >> 5;
  const int l15  = lane & 15, h = lane >> 4;
  const int wq   = wave >> 2, wr = wave & 3;        // 4x4 wave grid
  const int rowbase = by * 128;
  const int colbase = bx * 128;

  v8f acc[2][2] = {};

  // per-thread staging assignment: one 16B chunk per plane per K-step
  const int lrow = tid >> 2;                        // 0..127
  const int lch  = (tid & 3) * 8;                   // bf16 chunk (16 B)
  const int ldsi = lrow * 40 + lch;

  const uint64_t gAh = (uint64_t)(uintptr_t)(Zh + (size_t)(rowbase + lrow) * KZ + lch);
  const uint64_t gAl = (uint64_t)(uintptr_t)(Zl + (size_t)(rowbase + lrow) * KZ + lch);
  const uint64_t gBh = (uint64_t)(uintptr_t)(Zh + (size_t)(colbase + lrow) * KZ + lch);
  const uint64_t gBl = (uint64_t)(uintptr_t)(Zl + (size_t)(colbase + lrow) * KZ + lch);

  uint32_t lAh[2], lAl[2], lBh[2], lBl[2];
#pragma unroll
  for (int b = 0; b < 2; ++b) {
    lAh[b] = lds_off(&Ah[b][ldsi]);
    lAl[b] = lds_off(&Al[b][ldsi]);
    lBh[b] = lds_off(&Bh[b][ldsi]);
    lBl[b] = lds_off(&Bl[b][ldsi]);
  }

  // prologue: stage first K tile into buffer 0
  async_b128(lAh[0], gAh);
  async_b128(lAl[0], gAl);
  async_b128(lBh[0], gBh);
  async_b128(lBl[0], gBl);

  int buf = 0;
  for (int k0 = 0; k0 < KZ; k0 += 32, buf ^= 1) {
    wait_async0();                    // buffer `buf` resident in LDS (this wave)
    __syncthreads();                  // ... and visible to all waves; prev compute done

    if (k0 + 32 < KZ) {               // prefetch next tile into the other buffer
      uint64_t off = (uint64_t)((k0 + 32) * 2);
      async_b128(lAh[buf ^ 1], gAh + off);
      async_b128(lAl[buf ^ 1], gAl + off);
      async_b128(lBh[buf ^ 1], gBh + off);
      async_b128(lBl[buf ^ 1], gBl + off);
    }

    v16bf ahf[2], alf[2], bhf[2], blf[2];
    V16U u;
#pragma unroll
    for (int rg = 0; rg < 2; ++rg) {                // A 16x32: chunks h*8, 16+h*8
      int ar = (wq * 32 + rg * 16 + l15) * 40;
      u.h8[0] = *(const v8bf*)&Ah[buf][ar + h * 8];
      u.h8[1] = *(const v8bf*)&Ah[buf][ar + 16 + h * 8];
      ahf[rg] = u.v;
      u.h8[0] = *(const v8bf*)&Al[buf][ar + h * 8];
      u.h8[1] = *(const v8bf*)&Al[buf][ar + 16 + h * 8];
      alf[rg] = u.v;
    }
#pragma unroll
    for (int cg = 0; cg < 2; ++cg) {                // B 32x16: chunks h*8, 16+h*8
      int br = (wr * 32 + cg * 16 + l15) * 40;
      u.h8[0] = *(const v8bf*)&Bh[buf][br + h * 8];
      u.h8[1] = *(const v8bf*)&Bh[buf][br + 16 + h * 8];
      bhf[cg] = u.v;
      u.h8[0] = *(const v8bf*)&Bl[buf][br + h * 8];
      u.h8[1] = *(const v8bf*)&Bl[buf][br + 16 + h * 8];
      blf[cg] = u.v;
    }
#pragma unroll
    for (int rg = 0; rg < 2; ++rg)
#pragma unroll
      for (int cg = 0; cg < 2; ++cg) {
        acc[rg][cg] = __builtin_amdgcn_wmma_f32_16x16x32_bf16(
            false, ahf[rg], false, bhf[cg], (short)0, acc[rg][cg], false, false);
        acc[rg][cg] = __builtin_amdgcn_wmma_f32_16x16x32_bf16(
            false, ahf[rg], false, blf[cg], (short)0, acc[rg][cg], false, false);
        acc[rg][cg] = __builtin_amdgcn_wmma_f32_16x16x32_bf16(
            false, alf[rg], false, bhf[cg], (short)0, acc[rg][cg], false, false);
      }
  }

#pragma unroll
  for (int rg = 0; rg < 2; ++rg)
#pragma unroll
    for (int cg = 0; cg < 2; ++cg) {
      const int col   = colbase + wr * 32 + cg * 16 + l15;
      const int rbase = rowbase + wq * 32 + rg * 16 + h * 8;
      v8f a = acc[rg][cg];
#pragma unroll
      for (int v = 0; v < 8; ++v)                    // normal tile
        S[(size_t)(rbase + v) * NN + col] = a[v];
      if (bx != by) {                                // mirrored tile, contiguous
        size_t toff = (size_t)col * NN + rbase;      // 32B-aligned (rbase % 8 == 0)
        float4 x0; x0.x = a[0]; x0.y = a[1]; x0.z = a[2]; x0.w = a[3];
        float4 x1; x1.x = a[4]; x1.y = a[5]; x1.z = a[6]; x1.w = a[7];
        *(float4*)&S[toff]     = x0;
        *(float4*)&S[toff + 4] = x1;
      }
    }
}

// ---------------------------------------------------------------------------
// Kernel 4: per-row top-30 -> scatter into zeroed row. One block per row.
// Wave32 shfl_xor argmax (lowest-index tie-break, matches lax.top_k).
// ---------------------------------------------------------------------------
__global__ __launch_bounds__(256) void topk_kernel(
    const float* __restrict__ S, float* __restrict__ out)
{
  __shared__ float rowv[NN];
  __shared__ float wv[8];
  __shared__ int   wi[8];
  __shared__ float tv[KTOP];
  __shared__ int   ti[KTOP];

  const int t    = threadIdx.x;
  const int lane = t & 31;
  const int wid  = t >> 5;
  const int r    = blockIdx.x;
  const float* srow = S + (size_t)r * NN;

  for (int i = t; i < NN; i += 256) rowv[i] = srow[i];
  __syncthreads();

  for (int it = 0; it < KTOP; ++it) {
    float bv = -INFINITY; int bi = NN;
    for (int i = t; i < NN; i += 256) {
      float v = rowv[i];
      if (v > bv) { bv = v; bi = i; }               // ascending -> lowest idx kept
    }
#pragma unroll
    for (int m = 16; m > 0; m >>= 1) {
      float ov = __shfl_xor(bv, m, 32);
      int   oi = __shfl_xor(bi, m, 32);
      if (ov > bv || (ov == bv && oi < bi)) { bv = ov; bi = oi; }
    }
    if (lane == 0) { wv[wid] = bv; wi[wid] = bi; }
    __syncthreads();
    if (t == 0) {
      float best = wv[0]; int besti = wi[0];
#pragma unroll
      for (int w = 1; w < 8; ++w)
        if (wv[w] > best || (wv[w] == best && wi[w] < besti)) { best = wv[w]; besti = wi[w]; }
      tv[it] = best; ti[it] = besti;
      rowv[besti] = -INFINITY;
    }
    __syncthreads();
  }

  float* orow = out + (size_t)r * NN;
  for (int i = t; i < NN; i += 256) orow[i] = 0.0f;
  __syncthreads();
  if (t < KTOP) orow[ti[t]] = tv[t];
}

// ---------------------------------------------------------------------------
extern "C" void kernel_launch(void* const* d_in, const int* in_sizes, int n_in,
                              void* d_out, int out_size, void* d_ws, size_t ws_size,
                              hipStream_t stream) {
  (void)in_sizes; (void)n_in; (void)out_size; (void)ws_size;
  const float* x    = (const float*)d_in[0];   // (N, 256)
  const float* xo   = (const float*)d_in[1];   // (N, 600)
  const float* pw   = (const float*)d_in[2];   // (256, 256)
  const float* pb   = (const float*)d_in[3];   // (256,)
  const float* pow_ = (const float*)d_in[4];   // (600, 600)
  const float* pob  = (const float*)d_in[5];   // (600,)

  float* out    = (float*)d_out;               // (N, N) masked top-k
  float* scores = out + (size_t)NN * NN;       // (N, N) full scores

  __bf16* Zh = (__bf16*)d_ws;                  // N x 896 bf16 hi plane
  __bf16* Zl = Zh + (size_t)NN * KZ;           // N x 896 bf16 lo plane

  proj_x_kernel<<<NN / 16, 512, 0, stream>>>(x, pw, pb, Zh, Zl);
  proj_o_kernel<<<NN / 16, 512, 0, stream>>>(xo, pow_, pob, Zh, Zl);
  gram_kernel<<<NTRI, 512, 0, stream>>>(Zh, Zl, scores);
  topk_kernel<<<NN, 256, 0, stream>>>(scores, out);
}
